// CopyGenerator_11252814316328
// MI455X (gfx1250) — matrix operand order repaired and checked by reading, not compiled, not run on hardware
//
#include <hip/hip_runtime.h>
#include <hip/hip_bf16.h>
#include <stdint.h>

// ---------------- problem constants ----------------
#define TLEN    64
#define BATCH   32
#define SLEN    128
#define CVOCAB  64
#define DDIM    1024
#define VOCAB   50257
#define NROWS   (TLEN * BATCH)          // 2048
#define OUTW    (VOCAB + CVOCAB)        // 50321
#define VPAD    50304                   // 393 * 128, padded vocab for tiling

// ---------------- WMMA vector types ----------------
typedef __attribute__((ext_vector_type(16))) __bf16 v16bf;
typedef __attribute__((ext_vector_type(8)))  float  v8f;

union BFrag {
    v16bf v;      // 32 bytes = 8 VGPRs, bf16 WMMA operand
    uint4 u[2];   // two 16B global loads
};

// ---------------- f32 -> bf16 (RNE), with zero-padded tail ----------------
__global__ __launch_bounds__(256) void cvt_bf16_kernel(const float* __restrict__ src,
                                                       uint16_t* __restrict__ dst,
                                                       int nsrc, int ndst) {
    for (int i = blockIdx.x * 256 + threadIdx.x; i < ndst; i += gridDim.x * 256) {
        float f = (i < nsrc) ? src[i] : 0.0f;
        uint32_t u = __float_as_uint(f);
        u += 0x7FFFu + ((u >> 16) & 1u);          // round-to-nearest-even
        dst[i] = (uint16_t)(u >> 16);
    }
}

// ---------------- copy gate: p = sigmoid(hidden . W_copy + b_copy) ----------------
__global__ __launch_bounds__(256) void copy_gate_kernel(const float* __restrict__ hidden,
                                                        const float* __restrict__ Wc,
                                                        const float* __restrict__ bc,
                                                        float* __restrict__ pc) {
    __shared__ float red[256];
    const int row = blockIdx.x;
    float s = 0.0f;
    for (int i = threadIdx.x; i < DDIM; i += 256)
        s += hidden[(size_t)row * DDIM + i] * Wc[i];
    red[threadIdx.x] = s;
    __syncthreads();
    for (int off = 128; off > 0; off >>= 1) {
        if (threadIdx.x < off) red[threadIdx.x] += red[threadIdx.x + off];
        __syncthreads();
    }
    if (threadIdx.x == 0)
        pc[row] = 1.0f / (1.0f + __expf(-(red[0] + bc[0])));
}

// ---------------- WMMA GEMM: logits = A(bf16) x B(bf16)^T + bias ----------------
// Block = 8 waves: 4 (M) x 2 (N).  Block tile = 128(M) x 128(N).
// Wave tile = 32(M) x 64(N): 8 accumulators, per K-step 4 A-frag + 8 B-frag
// b128 loads feed 8 WMMAs (1.5 loads/WMMA; each B frag reused for 2 WMMAs).
__global__ __launch_bounds__(256) void gemm_wmma_kernel(const uint16_t* __restrict__ A,   // [NROWS, DDIM] bf16
                                                        const uint16_t* __restrict__ B,   // [VPAD , DDIM] bf16
                                                        const float* __restrict__ bias,   // [VOCAB]
                                                        float* __restrict__ out) {        // [NROWS, OUTW]
    const int tid  = threadIdx.x;
    const int lane = tid & 31;
    const int wave = tid >> 5;
    const int wm   = wave & 3;   // 0..3  -> M sub-tile (32 rows each)
    const int wn   = wave >> 2;  // 0..1  -> N half (64 cols each)

    const int rowBase = blockIdx.y * 128 + wm * 32;
    const int colBase = blockIdx.x * 128 + wn * 64;

    const int l16 = lane & 15;
    const int hi  = lane >> 4;           // 0 | 1
    const int qa  = hi << 3;             // A-frag chunks at qa and qa+16 (ISA 16-bit A 16x32 layout)
    const int qb  = hi << 3;             // B-frag chunks at qb and qb+16 (same K order as A)

    const uint16_t* aRow0 = A + (size_t)(rowBase + l16) * DDIM;        // rows rowBase..+15
    const uint16_t* aRow1 = aRow0 + (size_t)16 * DDIM;                 // rows rowBase+16..+31
    const uint16_t* bRow  = B + (size_t)(colBase + l16) * DDIM;

    v8f zero = {0.f, 0.f, 0.f, 0.f, 0.f, 0.f, 0.f, 0.f};
    v8f acc[2][4];
#pragma unroll
    for (int i = 0; i < 2; ++i)
#pragma unroll
        for (int j = 0; j < 4; ++j) acc[i][j] = zero;

#pragma unroll 2
    for (int k0 = 0; k0 < DDIM; k0 += 32) {
        BFrag fa0, fa1;
        fa0.u[0] = *(const uint4*)(aRow0 + k0 + qa);
        fa0.u[1] = *(const uint4*)(aRow0 + k0 + qa + 16);
        fa1.u[0] = *(const uint4*)(aRow1 + k0 + qa);
        fa1.u[1] = *(const uint4*)(aRow1 + k0 + qa + 16);
#pragma unroll
        for (int j = 0; j < 4; ++j) {
            BFrag fb;
            const uint16_t* bp = bRow + (size_t)j * 16 * DDIM + k0 + qb;
            fb.u[0] = *(const uint4*)(bp);
            fb.u[1] = *(const uint4*)(bp + 16);
            acc[0][j] = __builtin_amdgcn_wmma_f32_16x16x32_bf16(
                            false, fa0.v, false, fb.v, (short)0, acc[0][j], false, false);
            acc[1][j] = __builtin_amdgcn_wmma_f32_16x16x32_bf16(
                            false, fa1.v, false, fb.v, (short)0, acc[1][j], false, false);
        }
    }

    // C/D layout: lanes 0-15 -> N=lane, M=r; lanes 16-31 -> N=lane-16, M=8+r
    const int mBase = rowBase + (hi << 3);
#pragma unroll
    for (int j = 0; j < 4; ++j) {
        const int n = colBase + j * 16 + l16;
        if (n < VOCAB) {
            const float bv = bias[n];
#pragma unroll
            for (int i = 0; i < 2; ++i) {
                const int m0 = mBase + i * 16;
#pragma unroll
                for (int r = 0; r < 8; ++r) {
                    float v = acc[i][j][r] + bv;
                    if (n == 0) v = -3.0e38f;   // PAD_IDX masked to -inf
                    out[(size_t)(m0 + r) * OUTW + n] = v;
                }
            }
        }
    }
}

// ---------------- per-row online softmax stats (single read pass) ----------------
__global__ __launch_bounds__(256) void row_stats_kernel(const float* __restrict__ out,
                                                        float* __restrict__ rmax,
                                                        float* __restrict__ rsum) {
    __shared__ float sm[256], ss[256];
    const int row = blockIdx.x;
    const float* p = out + (size_t)row * OUTW;
    float m = -3.4e38f, s = 0.0f;
    for (int i = threadIdx.x; i < VOCAB; i += 256) {
        float v = p[i];
        if (v > m) { s = s * __expf(m - v) + 1.0f; m = v; }
        else       { s += __expf(v - m); }
    }
    sm[threadIdx.x] = m; ss[threadIdx.x] = s;
    __syncthreads();
    for (int off = 128; off > 0; off >>= 1) {
        if (threadIdx.x < off) {
            float m1 = sm[threadIdx.x],       s1 = ss[threadIdx.x];
            float m2 = sm[threadIdx.x + off], s2 = ss[threadIdx.x + off];
            float M = fmaxf(m1, m2);
            sm[threadIdx.x] = M;
            ss[threadIdx.x] = s1 * __expf(m1 - M) + s2 * __expf(m2 - M);
        }
        __syncthreads();
    }
    if (threadIdx.x == 0) { rmax[row] = sm[0]; rsum[row] = ss[0]; }
}

// ---------------- normalize: out = exp(l-m)/s * (1 - p_copy) ----------------
__global__ __launch_bounds__(256) void prob_finish_kernel(float* __restrict__ out,
                                                          const float* __restrict__ rmax,
                                                          const float* __restrict__ rsum,
                                                          const float* __restrict__ pc) {
    const int row = blockIdx.y;
    const int col = blockIdx.x * 256 + threadIdx.x;
    if (col >= VOCAB) return;
    const size_t idx = (size_t)row * OUTW + col;
    const float scale = (1.0f - pc[row]) / rsum[row];
    out[idx] = __expf(out[idx] - rmax[row]) * scale;
}

// ---------------- copy distribution: out[n, VOCAB+c] = p_copy * sum_s attn*src_map ----------------
__global__ __launch_bounds__(64) void copy_prob_kernel(const float* __restrict__ attn,
                                                       const float* __restrict__ smap,
                                                       const float* __restrict__ pc,
                                                       float* __restrict__ out) {
    const int n = blockIdx.x;         // tlen-major row: n = t*BATCH + b
    const int b = n & (BATCH - 1);
    const int c = threadIdx.x;        // 0..63
    float acc = 0.0f;
    for (int s = 0; s < SLEN; ++s)
        acc += attn[(size_t)n * SLEN + s] * smap[((size_t)s * BATCH + b) * CVOCAB + c];
    out[(size_t)n * OUTW + VOCAB + c] = acc * pc[n];
}

// ---------------- launcher ----------------
extern "C" void kernel_launch(void* const* d_in, const int* in_sizes, int n_in,
                              void* d_out, int out_size, void* d_ws, size_t ws_size,
                              hipStream_t stream) {
    const float* hidden = (const float*)d_in[0];   // [2048, 1024]
    const float* attn   = (const float*)d_in[1];   // [2048, 128]
    const float* smap   = (const float*)d_in[2];   // [128, 32, 64]
    const float* Wg     = (const float*)d_in[3];   // [50257, 1024]
    const float* bg     = (const float*)d_in[4];   // [50257]
    const float* Wc     = (const float*)d_in[5];   // [1, 1024]
    const float* bc     = (const float*)d_in[6];   // [1]
    float* out = (float*)d_out;                    // [2048, 50321]

    char* ws = (char*)d_ws;
    uint16_t* Abf = (uint16_t*)ws;                                     //   4 MB
    uint16_t* Bbf = (uint16_t*)(ws + (size_t)NROWS * DDIM * 2);        // 103 MB (fits L2 for GEMM reuse)
    float* rmax = (float*)(ws + (size_t)NROWS * DDIM * 2 + (size_t)VPAD * DDIM * 2);
    float* rsum = rmax + NROWS;
    float* pc   = rsum + NROWS;

    // 1) precision down-convert so W_gen (103 MB bf16) is L2-resident during the GEMM
    cvt_bf16_kernel<<<1024, 256, 0, stream>>>(hidden, Abf, NROWS * DDIM, NROWS * DDIM);
    cvt_bf16_kernel<<<8192, 256, 0, stream>>>(Wg, Bbf, VOCAB * DDIM, VPAD * DDIM);

    // 2) copy gate (independent of GEMM)
    copy_gate_kernel<<<NROWS, 256, 0, stream>>>(hidden, Wc, bc, pc);

    // 3) WMMA GEMM, logits written in place into the output buffer
    dim3 gg(VPAD / 128, NROWS / 128);
    gemm_wmma_kernel<<<gg, 256, 0, stream>>>(Abf, Bbf, bg, out);

    // 4) softmax stats, 5) normalize + gate
    row_stats_kernel<<<NROWS, 256, 0, stream>>>(out, rmax, rsum);
    dim3 gp((VOCAB + 255) / 256, NROWS);
    prob_finish_kernel<<<gp, 256, 0, stream>>>(out, rmax, rsum, pc);

    // 6) copy-attention distribution into the extended-vocab columns
    copy_prob_kernel<<<NROWS, 64, 0, stream>>>(attn, smap, pc, out);
}
